// SBGAD_67053029425341
// MI455X (gfx1250) — matrix operand order repaired and checked by reading, not compiled, not run on hardware
//
#include <hip/hip_runtime.h>
#include <hip/hip_bf16.h>

#define B_ 8
#define CIN 256
#define Hh 256
#define Ww 256
#define HW 65536
#define NSEL 6554            // int(65536*0.1)+1
#define TOT (B_ * NSEL)      // 52432
#define TOT64 (TOT * 64)
#define C0_IDX 917
#define C1_IDX 31337

typedef __bf16 v16bf __attribute__((ext_vector_type(16)));
typedef float  v8f   __attribute__((ext_vector_type(8)));
typedef unsigned int u32x4 __attribute__((ext_vector_type(4)));
typedef int i32x8 __attribute__((ext_vector_type(8)));
typedef int i32x4 __attribute__((ext_vector_type(4)));

// ---------------- scan helper (wave32-safe, LDS based) ----------------
__device__ inline int block_scan_incl(int v, int* sm) {
  int t = threadIdx.x;
  sm[t] = v; __syncthreads();
  for (int off = 1; off < 256; off <<= 1) {
    int x = (t >= off) ? sm[t - off] : 0;
    __syncthreads();
    sm[t] += x;
    __syncthreads();
  }
  return sm[t];
}

// ---------------- WMMA fragment loaders (ISA 7.12.2 layouts) ----------------
// A: 16x32 bf16.  lane<16 holds M=lane: v0..3 -> K 0..7, v4..7 -> K16..23;
// lanes 16..31 hold same M, K offset +8 (and +8 in upper half).
__device__ inline v16bf load_a_bf16(const __bf16* X, int ld, int m0, int k0, int lane) {
  int m = m0 + (lane & 15), hi = lane >> 4;
  v16bf a;
#pragma unroll
  for (int v = 0; v < 8; ++v) {
    int kb = k0 + hi * 8 + (v < 4 ? 2 * v : 16 + 2 * (v - 4));
    a[2 * v]     = X[m * ld + kb];
    a[2 * v + 1] = X[m * ld + kb + 1];
  }
  return a;
}
__device__ inline v16bf load_a_f32(const float* X, int ld, int m0, int k0, int lane) {
  int m = m0 + (lane & 15), hi = lane >> 4;
  v16bf a;
#pragma unroll
  for (int v = 0; v < 8; ++v) {
    int kb = k0 + hi * 8 + (v < 4 ? 2 * v : 16 + 2 * (v - 4));
    a[2 * v]     = (__bf16)X[m * ld + kb];
    a[2 * v + 1] = (__bf16)X[m * ld + kb + 1];
  }
  return a;
}
// B: 32x16 bf16, B[k,n] = W[n,k] (W row-major [N][K], ld = K stride).
// lane<16 holds N=lane: v0..3 -> K 0..7, v4..7 -> K16..23; lanes 16..31: K offset +8.
__device__ inline v16bf load_b_w(const __bf16* W, int ld, int n0, int k0, int lane) {
  int n = n0 + (lane & 15), hi = lane >> 4;
  v16bf b;
#pragma unroll
  for (int v = 0; v < 8; ++v) {
    int k = k0 + hi * 8 + (v < 4 ? 2 * v : 16 + 2 * (v - 4));
    b[2 * v]     = W[n * ld + k];
    b[2 * v + 1] = W[n * ld + k + 1];
  }
  return b;
}
__device__ inline v8f wmma_bf16(v16bf a, v16bf b, v8f c) {
  return __builtin_amdgcn_wmma_f32_16x16x32_bf16(false, a, false, b, (short)0, c, false, false);
}

// ---------------- 0: weights fp32 -> bf16 ----------------
__global__ void k_wconv(const float* w1, const float* w2, const float* dw1, const float* dw2,
                        __bf16* w1b, __bf16* w2b, __bf16* dw1b, __bf16* dw2b) {
  int i = blockIdx.x * 256 + threadIdx.x;
  if (i < 32768) w1b[i] = (__bf16)w1[i];
  if (i < 8192) { w2b[i] = (__bf16)w2[i]; dw1b[i] = (__bf16)dw1[i]; dw2b[i] = (__bf16)dw2[i]; }
}

// ---------------- 1: boundary map (memory-bound streaming) ----------------
__global__ __launch_bounds__(256) void k_bnd(const float* __restrict__ Fe, float* __restrict__ bnd) {
  int b = blockIdx.y;
  int y0 = blockIdx.x * 8;
  int x = threadIdx.x;
  __shared__ float rows[9][256];
  float acc[8];
#pragma unroll
  for (int r = 0; r < 8; ++r) acc[r] = 0.f;
  const float* base = Fe + (size_t)b * CIN * HW;
  for (int c = 0; c < CIN; ++c) {
    const float* p = base + (size_t)c * HW + y0 * Ww + x;
    __builtin_prefetch(p + HW, 0, 0);   // global_prefetch_b8: next channel plane
#pragma unroll
    for (int r = 0; r < 9; ++r) {
      rows[r][x] = (y0 + r < Hh) ? p[r * Ww] : 0.f;
    }
    __syncthreads();
#pragma unroll
    for (int r = 0; r < 8; ++r) {
      float v = rows[r][x];
      float gx = (x < Ww - 1) ? fabsf(rows[r][x + 1] - v) : fabsf(v - rows[r][Ww - 2]);
      int yy = y0 + r;
      float gy = (yy < Hh - 1) ? fabsf(rows[r + 1][x] - v) : fabsf(v - rows[r - 1][x]);
      acc[r] += gx + gy;
    }
    __syncthreads();
  }
#pragma unroll
  for (int r = 0; r < 8; ++r) bnd[(size_t)b * HW + (y0 + r) * Ww + x] = acc[r];
}

// ---------------- 2: top-N_SEL selection (exact count, index-sorted) ----------------
__global__ void k_hist(const float* __restrict__ bnd, unsigned int* __restrict__ hist) {
  int i = blockIdx.x * 256 + threadIdx.x;      // B*HW threads
  int b = i >> 16;
  unsigned key = __float_as_uint(bnd[i]);      // bnd >= 0 -> bit pattern monotone
  atomicAdd(&hist[(b << 16) + (key >> 16)], 1u);
}

__global__ void k_thresh(const unsigned int* __restrict__ hist, int* tbin, int* need_eq) {
  int b = blockIdx.x, t = threadIdx.x;
  __shared__ unsigned int csum[256];
  const unsigned int* h = hist + (b << 16);
  unsigned s = 0;
  for (int i = 0; i < 256; ++i) s += h[t * 256 + i];
  csum[t] = s; __syncthreads();
  if (t == 0) {
    unsigned acc = 0;
    int chunk = 255;
    for (; chunk > 0; --chunk) { if (acc + csum[chunk] >= (unsigned)NSEL) break; acc += csum[chunk]; }
    int bin = chunk * 256 + 255;
    for (; bin > chunk * 256; --bin) { if (acc + h[bin] >= (unsigned)NSEL) break; acc += h[bin]; }
    tbin[b] = bin;
    need_eq[b] = NSEL - (int)acc;   // ties to take from threshold bin, in index order
  }
}

__global__ void k_cutoff(const float* __restrict__ bnd, const int* tbin, const int* need_eq,
                         int* cutoff) {
  int b = blockIdx.x, t = threadIdx.x;
  __shared__ int sm[256];
  __shared__ int done, run;
  if (t == 0) { done = (need_eq[b] == 0); run = 0; cutoff[b] = -1; }
  __syncthreads();
  int tb = tbin[b], need = need_eq[b];
  for (int base = 0; base < HW; base += 256) {
    if (done) break;
    unsigned key = __float_as_uint(bnd[(size_t)b * HW + base + t]);
    int pred = ((int)(key >> 16) == tb);
    int inc = block_scan_incl(pred, sm);
    int tot = sm[255];
    if (pred && run + inc == need) cutoff[b] = base + t;
    __syncthreads();
    if (t == 0) { if (run + tot >= need) done = 1; else run += tot; }
    __syncthreads();
  }
}

__device__ inline int sel_pred(const float* bnd, const int* tbin, const int* cutoff,
                               int b, int pix) {
  int bin = (int)(__float_as_uint(bnd[(size_t)b * HW + pix]) >> 16);
  return (bin > tbin[b]) || (bin == tbin[b] && pix <= cutoff[b]);
}

__global__ void k_count(const float* __restrict__ bnd, const int* tbin, const int* cutoff,
                        int* blkcnt) {
  int blk = blockIdx.x;                 // 2048 blocks, 256 px each
  int b = blk >> 8;
  int pix = ((blk & 255) << 8) + threadIdx.x;
  __shared__ int sm[256];
  int incl = block_scan_incl(sel_pred(bnd, tbin, cutoff, b, pix), sm);
  if (threadIdx.x == 255) blkcnt[blk] = incl;
}

__global__ void k_scanblk(const int* __restrict__ blkcnt, int* __restrict__ blkoff) {
  int b = threadIdx.x;
  if (b < B_) {
    int acc = 0;
    for (int i = 0; i < 256; ++i) { blkoff[b * 256 + i] = acc; acc += blkcnt[b * 256 + i]; }
  }
}

__global__ void k_write(const float* __restrict__ bnd, const int* tbin, const int* cutoff,
                        const int* __restrict__ blkoff, int* __restrict__ idx_sel) {
  int blk = blockIdx.x;
  int b = blk >> 8;
  int pix = ((blk & 255) << 8) + threadIdx.x;
  __shared__ int sm[256];
  int sel = sel_pred(bnd, tbin, cutoff, b, pix);
  int incl = block_scan_incl(sel, sm);
  if (sel) idx_sel[b * NSEL + blkoff[blk] + incl - 1] = pix;
}

// ---------------- 3: projection MLP at selected pixels (WMMA bf16) ----------------
__global__ __launch_bounds__(256) void k_project(const float* __restrict__ Fe,
                                                 const int* __restrict__ idx_sel,
                                                 const __bf16* __restrict__ w1b,
                                                 const __bf16* __restrict__ w2b,
                                                 const float* __restrict__ b1,
                                                 const float* __restrict__ b2,
                                                 float* __restrict__ Mk) {
  __shared__ __bf16 Xt[64 * 264];     // 64 rows x 256 ch (pad 8)
  __shared__ __bf16 Ht[64 * 136];     // 64 rows x 128 (pad 8)
  int row0 = blockIdx.x * 64;
  int t = threadIdx.x;
  // gather X (scattered channel reads) -> bf16 LDS
  for (int i = t; i < 64 * 256; i += 256) {
    int r = i >> 8, c = i & 255;
    int rg = row0 + r;
    float v = 0.f;
    if (rg < TOT) {
      int b = rg / NSEL;
      int pix = idx_sel[rg];
      v = Fe[((size_t)b * CIN + c) * HW + pix];
    }
    Xt[r * 264 + c] = (__bf16)v;
  }
  __syncthreads();

  int wave = t >> 5, lane = t & 31;
  int nn = lane & 15, hi = lane >> 4;

  // GEMM1: H[64x128] = relu(X[64x256] @ w1^T + b1); wave -> N-tile, 4 M-tiles each
  {
    v8f acc[4];
#pragma unroll
    for (int m = 0; m < 4; ++m)
#pragma unroll
      for (int j = 0; j < 8; ++j) acc[m][j] = 0.f;
    int n0 = wave * 16;
    for (int ks = 0; ks < 8; ++ks) {
      int k0 = ks * 32;
      v16bf bf = load_b_w(w1b, 256, n0, k0, lane);
#pragma unroll
      for (int mt = 0; mt < 4; ++mt) {
        v16bf af = load_a_bf16(Xt, 264, mt * 16, k0, lane);
        acc[mt] = wmma_bf16(af, bf, acc[mt]);
      }
    }
    float bias = b1[n0 + nn];
#pragma unroll
    for (int mt = 0; mt < 4; ++mt)
#pragma unroll
      for (int r = 0; r < 8; ++r) {
        int m = mt * 16 + hi * 8 + r;
        float v = acc[mt][r] + bias;
        Ht[m * 136 + n0 + nn] = (__bf16)(v > 0.f ? v : 0.f);
      }
  }
  __syncthreads();

  // GEMM2: A[64x64] = H @ w2^T + b2; 16 tiles over 8 waves -> 2 tiles each
  {
    v8f acc[2];
#pragma unroll
    for (int m = 0; m < 2; ++m)
#pragma unroll
      for (int j = 0; j < 8; ++j) acc[m][j] = 0.f;
    int n0 = (wave & 3) * 16;
    int mbase = (wave >> 2) * 32;
    for (int ks = 0; ks < 4; ++ks) {
      int k0 = ks * 32;
      v16bf bf = load_b_w(w2b, 128, n0, k0, lane);
#pragma unroll
      for (int mt = 0; mt < 2; ++mt) {
        v16bf af = load_a_bf16(Ht, 136, mbase + mt * 16, k0, lane);
        acc[mt] = wmma_bf16(af, bf, acc[mt]);
      }
    }
    float bias = b2[n0 + nn];
#pragma unroll
    for (int mt = 0; mt < 2; ++mt)
#pragma unroll
      for (int r = 0; r < 8; ++r) {
        int m = mbase + mt * 16 + hi * 8 + r;
        int rg = row0 + m;
        if (rg < TOT) Mk[(size_t)rg * 64 + n0 + nn] = acc[mt][r] + bias;
      }
  }
}

// ---------------- 4: centers / labels / partition ----------------
__global__ void k_centers(const float* __restrict__ Mk, float* __restrict__ centers) {
  int t = threadIdx.x;   // 128 threads
  if (t < 128) {
    int src = (t < 64) ? C0_IDX : C1_IDX;
    centers[t] = Mk[(size_t)src * 64 + (t & 63)];
  }
}

__global__ void k_labels(const float* __restrict__ Mk, const float* __restrict__ centers,
                         int* __restrict__ labels) {
  int rg = blockIdx.x * 256 + threadIdx.x;
  if (rg >= TOT) return;
  const float* x = Mk + (size_t)rg * 64;
  float d0 = 0.f, d1 = 0.f;
#pragma unroll 8
  for (int a = 0; a < 64; ++a) {
    float v = x[a];
    float e0 = v - centers[a], e1 = v - centers[64 + a];
    d0 += e0 * e0; d1 += e1 * e1;
  }
  labels[rg] = (d1 < d0) ? 1 : 0;
}

__global__ void k_partition(const int* __restrict__ labels, int* __restrict__ ord0,
                            int* __restrict__ ord1, int* __restrict__ cnt) {
  int b = blockIdx.x, t = threadIdx.x;
  __shared__ int sm[256];
  __shared__ int run0, run1;
  if (t == 0) { run0 = 0; run1 = 0; }
  __syncthreads();
  for (int base = 0; base < NSEL; base += 256) {
    int j = base + t;
    int lab = (j < NSEL) ? labels[b * NSEL + j] : -1;
    int p0 = (lab == 0);
    int s0 = block_scan_incl(p0, sm);
    int t0 = sm[255];
    if (p0) ord0[b * NSEL + run0 + s0 - 1] = j;
    __syncthreads();
    int p1 = (lab == 1);
    int s1 = block_scan_incl(p1, sm);
    int t1 = sm[255];
    if (p1) ord1[b * NSEL + run1 + s1 - 1] = j;
    __syncthreads();
    if (t == 0) { run0 += t0; run1 += t1; }
    __syncthreads();
  }
  if (t == 0) { cnt[b] = run0; cnt[B_ + b] = run1; }
}

__global__ void k_msinfo(const int* __restrict__ cnt, int* __restrict__ msout) {
  if (threadIdx.x == 0) {
    int L1 = 0, L2 = 0;
    for (int b = 0; b < B_; ++b) {
      L1 = (cnt[b] > L1) ? cnt[b] : L1;
      L2 = (cnt[B_ + b] > L2) ? cnt[B_ + b] : L2;
    }
    msout[0] = (L1 < L2) ? L1 : L2;
  }
}

__global__ void k_buildA(const float* __restrict__ Mk, const int* __restrict__ ord0,
                         const int* __restrict__ ord1, const int* __restrict__ cnt,
                         float* __restrict__ A1, float* __restrict__ A2) {
  int i = blockIdx.x * 256 + threadIdx.x;
  if (i >= TOT64) return;
  int rg = i >> 6, a = i & 63;
  int b = rg / NSEL, pos = rg % NSEL;
  float v1 = 0.f, v2 = 0.f;
  if (pos < cnt[b])      v1 = Mk[((size_t)b * NSEL + ord0[b * NSEL + pos]) * 64 + a];
  if (pos < cnt[B_ + b]) v2 = Mk[((size_t)b * NSEL + ord1[b * NSEL + pos]) * 64 + a];
  A1[i] = v1; A2[i] = v2;
}

// ---------------- 5: decoder MLP (TDM-staged input, WMMA bf16) ----------------
__global__ __launch_bounds__(256) void k_decoder(const float* __restrict__ Xin,
                                                 const __bf16* __restrict__ dw1b,
                                                 const __bf16* __restrict__ dw2b,
                                                 const float* __restrict__ db1,
                                                 const float* __restrict__ db2,
                                                 float* __restrict__ Yout) {
  __shared__ float  Xf[64 * 72];     // TDM pads 8 DWORDs after every 64 -> ld 72
  __shared__ __bf16 Ht[64 * 136];
  int row0 = blockIdx.x * 64;
  int t = threadIdx.x;

  if (t < 32) {   // wave 0 issues the Tensor-Data-Mover load (EXEC-independent DMA)
    unsigned long long ga = (unsigned long long)(uintptr_t)(Xin + (size_t)row0 * 64);
    unsigned int lds = (unsigned int)(uintptr_t)&Xf[0];    // low 32 bits = LDS byte offset
    unsigned int rows_rem = (unsigned int)(TOT - row0);    // OOB rows read as zero
    u32x4 g0; i32x8 g1; i32x4 g2, g3; i32x8 g4;
    g0[0] = 1u;                                  // count=1, user descriptor
    g0[1] = lds;                                 // lds_addr
    g0[2] = (unsigned int)(ga & 0xffffffffu);    // global_addr[31:0]
    g0[3] = (unsigned int)((ga >> 32) & 0x1ffffffu) | (2u << 30);  // addr[56:32] | type=2
    g1[0] = (int)((2u << 16) | (1u << 20) | (5u << 22) | (7u << 25)); // 4B elems, pad 8dw/64dw
    g1[1] = (int)(64u << 16);                    // tensor_dim0 = 64
    g1[2] = (int)((rows_rem & 0xffffu) << 16);   // tensor_dim1 lo
    g1[3] = (int)(((rows_rem >> 16) & 0xffffu) | (64u << 16)); // dim1 hi | tile_dim0=64
    g1[4] = 64;                                  // tile_dim1 = 64, tile_dim2 = 0
    g1[5] = 64;                                  // tensor_dim0_stride = 64
    g1[6] = 0; g1[7] = 0;
    g2[0] = g2[1] = g2[2] = g2[3] = 0;
    g3[0] = g3[1] = g3[2] = g3[3] = 0;
    g4[0] = g4[1] = g4[2] = g4[3] = 0;
    g4[4] = g4[5] = g4[6] = g4[7] = 0;
    __builtin_amdgcn_tensor_load_to_lds(g0, g1, g2, g3, g4, 0);
    __builtin_amdgcn_s_wait_tensorcnt(0);
  }
  __syncthreads();

  int wave = t >> 5, lane = t & 31;
  int nn = lane & 15, hi = lane >> 4;

  // GEMM1: H[64x128] = relu(X[64x64] @ dw1^T + db1), K=64 (2 k-steps)
  {
    v8f acc[4];
#pragma unroll
    for (int m = 0; m < 4; ++m)
#pragma unroll
      for (int j = 0; j < 8; ++j) acc[m][j] = 0.f;
    int n0 = wave * 16;
    for (int ks = 0; ks < 2; ++ks) {
      int k0 = ks * 32;
      v16bf bf = load_b_w(dw1b, 64, n0, k0, lane);
#pragma unroll
      for (int mt = 0; mt < 4; ++mt) {
        v16bf af = load_a_f32(Xf, 72, mt * 16, k0, lane);
        acc[mt] = wmma_bf16(af, bf, acc[mt]);
      }
    }
    float bias = db1[n0 + nn];
#pragma unroll
    for (int mt = 0; mt < 4; ++mt)
#pragma unroll
      for (int r = 0; r < 8; ++r) {
        int m = mt * 16 + hi * 8 + r;
        float v = acc[mt][r] + bias;
        Ht[m * 136 + n0 + nn] = (__bf16)(v > 0.f ? v : 0.f);
      }
  }
  __syncthreads();

  // GEMM2: Y[64x64] = H @ dw2^T + db2, K=128 (4 k-steps)
  {
    v8f acc[2];
#pragma unroll
    for (int m = 0; m < 2; ++m)
#pragma unroll
      for (int j = 0; j < 8; ++j) acc[m][j] = 0.f;
    int n0 = (wave & 3) * 16;
    int mbase = (wave >> 2) * 32;
    for (int ks = 0; ks < 4; ++ks) {
      int k0 = ks * 32;
      v16bf bf = load_b_w(dw2b, 128, n0, k0, lane);
#pragma unroll
      for (int mt = 0; mt < 2; ++mt) {
        v16bf af = load_a_bf16(Ht, 136, mbase + mt * 16, k0, lane);
        acc[mt] = wmma_bf16(af, bf, acc[mt]);
      }
    }
    float bias = db2[n0 + nn];
#pragma unroll
    for (int mt = 0; mt < 2; ++mt)
#pragma unroll
      for (int r = 0; r < 8; ++r) {
        int m = mbase + mt * 16 + hi * 8 + r;
        int rg = row0 + m;
        if (rg < TOT) Yout[(size_t)rg * 64 + n0 + nn] = acc[mt][r] + bias;
      }
  }
}

// ---------------- 6: masked L1 loss reduction ----------------
__global__ void k_loss(const float* __restrict__ A1, const float* __restrict__ A1p,
                       const float* __restrict__ A2, const float* __restrict__ A2p,
                       const int* __restrict__ msp, float* __restrict__ partial) {
  int ms = msp[0];
  int i = blockIdx.x * 256 + threadIdx.x;
  float v = 0.f;
  if (i < TOT64) {
    int rg = i >> 6;
    int pos = rg % NSEL;
    if (pos < ms) v = fabsf(A1p[i] - A1[i]) + fabsf(A2p[i] - A2[i]);
  }
  __shared__ float sm[256];
  sm[threadIdx.x] = v; __syncthreads();
  for (int off = 128; off > 0; off >>= 1) {
    if (threadIdx.x < off) sm[threadIdx.x] += sm[threadIdx.x + off];
    __syncthreads();
  }
  if (threadIdx.x == 0) atomicAdd(partial, sm[0]);
}

__global__ void k_final(const float* __restrict__ partial, const int* __restrict__ msp,
                        float* __restrict__ out) {
  if (threadIdx.x == 0) out[0] = partial[0] / ((float)(B_ * 64) * (float)msp[0]);
}

// ---------------- launch ----------------
extern "C" void kernel_launch(void* const* d_in, const int* in_sizes, int n_in,
                              void* d_out, int out_size, void* d_ws, size_t ws_size,
                              hipStream_t stream) {
  const float* Fe  = (const float*)d_in[0];
  const float* w1  = (const float*)d_in[2];
  const float* b1  = (const float*)d_in[3];
  const float* w2  = (const float*)d_in[4];
  const float* b2  = (const float*)d_in[5];
  const float* dw1 = (const float*)d_in[6];
  const float* db1 = (const float*)d_in[7];
  const float* dw2 = (const float*)d_in[8];
  const float* db2 = (const float*)d_in[9];
  float* out = (float*)d_out;

  char* ws = (char*)d_ws;
  size_t off = 0;
  auto alloc = [&](size_t bytes) -> char* {
    char* p = ws + off;
    off = (off + bytes + 255) & ~(size_t)255;
    return p;
  };
  float*        bnd     = (float*)alloc((size_t)B_ * HW * 4);
  unsigned int* hist    = (unsigned int*)alloc((size_t)B_ * 65536 * 4);
  int*          tbin    = (int*)alloc(B_ * 4);
  int*          need_eq = (int*)alloc(B_ * 4);
  int*          cutoff  = (int*)alloc(B_ * 4);
  int*          cnt     = (int*)alloc(2 * B_ * 4);
  int*          msbuf   = (int*)alloc(4);
  float*        partial = (float*)alloc(4);
  int*          blkcnt  = (int*)alloc(2048 * 4);
  int*          blkoff  = (int*)alloc(2048 * 4);
  int*          idx_sel = (int*)alloc((size_t)TOT * 4);
  float*        Mk      = (float*)alloc((size_t)TOT64 * 4);
  int*          labels  = (int*)alloc((size_t)TOT * 4);
  int*          ord0    = (int*)alloc((size_t)TOT * 4);
  int*          ord1    = (int*)alloc((size_t)TOT * 4);
  float*        centers = (float*)alloc(128 * 4);
  float*        A1      = (float*)alloc((size_t)TOT64 * 4);
  float*        A2      = (float*)alloc((size_t)TOT64 * 4);
  float*        A1p     = (float*)alloc((size_t)TOT64 * 4);
  float*        A2p     = (float*)alloc((size_t)TOT64 * 4);
  __bf16*       w1b     = (__bf16*)alloc(32768 * 2);
  __bf16*       w2b     = (__bf16*)alloc(8192 * 2);
  __bf16*       dw1b    = (__bf16*)alloc(8192 * 2);
  __bf16*       dw2b    = (__bf16*)alloc(8192 * 2);

  (void)hipMemsetAsync(hist, 0, (size_t)B_ * 65536 * 4, stream);
  (void)hipMemsetAsync(partial, 0, 4, stream);

  k_wconv<<<128, 256, 0, stream>>>(w1, w2, dw1, dw2, w1b, w2b, dw1b, dw2b);
  k_bnd<<<dim3(Hh / 8, B_), 256, 0, stream>>>(Fe, bnd);
  k_hist<<<(B_ * HW) / 256, 256, 0, stream>>>(bnd, hist);
  k_thresh<<<B_, 256, 0, stream>>>(hist, tbin, need_eq);
  k_cutoff<<<B_, 256, 0, stream>>>(bnd, tbin, need_eq, cutoff);
  k_count<<<2048, 256, 0, stream>>>(bnd, tbin, cutoff, blkcnt);
  k_scanblk<<<1, 256, 0, stream>>>(blkcnt, blkoff);
  k_write<<<2048, 256, 0, stream>>>(bnd, tbin, cutoff, blkoff, idx_sel);

  int gemm_blocks = (TOT + 63) / 64;   // 820
  k_project<<<gemm_blocks, 256, 0, stream>>>(Fe, idx_sel, w1b, w2b, b1, b2, Mk);

  k_centers<<<1, 128, 0, stream>>>(Mk, centers);
  k_labels<<<(TOT + 255) / 256, 256, 0, stream>>>(Mk, centers, labels);
  k_partition<<<B_, 256, 0, stream>>>(labels, ord0, ord1, cnt);
  k_msinfo<<<1, 32, 0, stream>>>(cnt, msbuf);
  k_buildA<<<(TOT64 + 255) / 256, 256, 0, stream>>>(Mk, ord0, ord1, cnt, A1, A2);

  k_decoder<<<gemm_blocks, 256, 0, stream>>>(A2, dw1b, dw2b, db1, db2, A1p);
  k_decoder<<<gemm_blocks, 256, 0, stream>>>(A1, dw1b, dw2b, db1, db2, A2p);

  k_loss<<<(TOT64 + 255) / 256, 256, 0, stream>>>(A1, A1p, A2, A2p, msbuf, partial);
  k_final<<<1, 32, 0, stream>>>(partial, msbuf, out);

  (void)in_sizes; (void)n_in; (void)out_size; (void)ws_size;
}
